// BiLSTM_CRF_39316130627601
// MI455X (gfx1250) — compile-verified, matrix operand order checked by reading, not run on hardware
//
#include <hip/hip_runtime.h>

#define T_LEN 2048
#define IN_DIM 1024
#define HHD 512
#define G4 2048           // 4*HHD
#define NTAG 16
#define START_TAG 14
#define STOP_TAG 15
#define NEGV -10000.0f

typedef __attribute__((ext_vector_type(16))) _Float16 v16h;
typedef __attribute__((ext_vector_type(8)))  _Float16 v8h;
typedef __attribute__((ext_vector_type(8)))  float    v8f;
typedef __attribute__((ext_vector_type(4)))  float    v4f;

// ---------------------------------------------------------------------------
// Fragment loader: 16x32 f16 tile from a row-major matrix (ld = row stride in
// elements). Per CDNA5 ISA 16-bit A/B layout: lane L holds row (L&15),
// halves [koff..koff+7] then [koff+16..koff+23], koff = (L>>4)*8.
// Two 16-byte loads per lane.
// ---------------------------------------------------------------------------
__device__ __forceinline__ v16h load_frag16(const _Float16* __restrict__ base, int ld) {
  const int lane = threadIdx.x & 31;
  const int r    = lane & 15;
  const int koff = (lane >> 4) << 3;  // 0 or 8
  const _Float16* p = base + (size_t)r * ld + koff;
  union { v16h v; v8h h[2]; } u;
  u.h[0] = *(const v8h*)(p);
  u.h[1] = *(const v8h*)(p + 16);
  return u.v;
}

// ---------------------------------------------------------------------------
// f32 -> f16 elementwise conversion / bias sum / counter reset
// ---------------------------------------------------------------------------
__global__ void cvt_f32_f16(const float* __restrict__ src, _Float16* __restrict__ dst, int n) {
  int i = blockIdx.x * blockDim.x + threadIdx.x;
  if (i < n) dst[i] = (_Float16)src[i];
}

__global__ void add_vec(const float* __restrict__ a, const float* __restrict__ b,
                        float* __restrict__ o, int n) {
  int i = blockIdx.x * blockDim.x + threadIdx.x;
  if (i < n) o[i] = a[i] + b[i];
}

__global__ void zero_ints(int* __restrict__ p, int n) {
  int i = blockIdx.x * blockDim.x + threadIdx.x;
  if (i < n) p[i] = 0;
}

// ---------------------------------------------------------------------------
// WMMA GEMM: G[dir] = X[T,IN] @ W[dir][G4,IN]^T + bias[dir]   (f16 in, f32 out)
// One wave per block; each wave computes a 64x32 output block (4x2 tiles):
// 8 WMMAs per K-step against 6 fragment loads.  ~140 VGPRs -> no spills;
// __launch_bounds__(32,1) lets the allocator budget for a single-wave block.
// grid = (G4/32, T/64, 2), block = 32.
// ---------------------------------------------------------------------------
__global__ void __launch_bounds__(32, 1)
wmma_gemm_gates(const _Float16* __restrict__ X,
                const _Float16* __restrict__ Wf,
                const _Float16* __restrict__ Wb,
                const float* __restrict__ biasf,
                const float* __restrict__ biasb,
                float* __restrict__ Gf,
                float* __restrict__ Gb) {
  const int n0  = blockIdx.x * 32;
  const int m0  = blockIdx.y * 64;
  const int dir = blockIdx.z;
  const _Float16* W   = dir ? Wb : Wf;
  const float*    bia = dir ? biasb : biasf;
  float*          G   = dir ? Gb : Gf;

  v8f acc[4][2];
#pragma unroll
  for (int i = 0; i < 4; ++i)
#pragma unroll
    for (int j = 0; j < 2; ++j) acc[i][j] = (v8f){};

  for (int k0 = 0; k0 < IN_DIM; k0 += 32) {
    v16h a[4], b[2];
#pragma unroll
    for (int i = 0; i < 4; ++i)
      a[i] = load_frag16(X + (size_t)(m0 + 16 * i) * IN_DIM + k0, IN_DIM);
#pragma unroll
    for (int j = 0; j < 2; ++j)
      b[j] = load_frag16(W + (size_t)(n0 + 16 * j) * IN_DIM + k0, IN_DIM);
#pragma unroll
    for (int i = 0; i < 4; ++i)
#pragma unroll
      for (int j = 0; j < 2; ++j)
        acc[i][j] = __builtin_amdgcn_wmma_f32_16x16x32_f16(false, a[i], false, b[j],
                                                           (short)0, acc[i][j], false, false);
  }

  // D layout: lane (0-15) -> N=lane, VGPR r -> M=r; lanes 16-31 -> M=r+8
  const int lane = threadIdx.x & 31;
  const int colL = lane & 15;
  const int rhi  = (lane >> 4) * 8;
  float bc[2];
#pragma unroll
  for (int j = 0; j < 2; ++j) bc[j] = bia[n0 + 16 * j + colL];
#pragma unroll
  for (int i = 0; i < 4; ++i) {
#pragma unroll
    for (int r = 0; r < 8; ++r) {
      const int row = m0 + 16 * i + r + rhi;
#pragma unroll
      for (int j = 0; j < 2; ++j) {
        const int col = n0 + 16 * j + colL;
        G[(size_t)row * G4 + col] = acc[i][j][r] + bc[j];
      }
    }
  }
}

// ---------------------------------------------------------------------------
// Serial bidirectional LSTM scan, split across 4 WGPs per direction.
// grid = 8 (dir = blk>>2, part = blk&3), block = 512 (16 waves).
// Each block owns 128 hidden units => 512 gate rows of W_hh (f16, ~512KB,
// L2-resident).  h (f16) is double-buffered in global; per-step cross-WGP
// sync uses a monotonically increasing release/acquire counter per direction.
// Per step: stage h into LDS (1KB), 512-element packed-f16 dot per thread
// (v_pk_fma_f16), gate exchange through LDS, c/h update on threads 0..127.
// ---------------------------------------------------------------------------
__global__ void lstm_scan_mb(const float* __restrict__ Gf,
                             const float* __restrict__ Gb,
                             const _Float16* __restrict__ Whhf16, // [G4, HHD]
                             const _Float16* __restrict__ Whhb16,
                             const float* __restrict__ h0,        // [2,1,HHD]
                             const float* __restrict__ c0,
                             _Float16* __restrict__ hbuf,         // [2][2][HHD] f16
                             int* __restrict__ cnt,               // [2]
                             _Float16* __restrict__ H)            // [T, 2*HHD]
{
  __shared__ __align__(16) _Float16 sH[HHD];
  __shared__ __align__(16) float    sG[HHD];

  const int dir  = blockIdx.x >> 2;
  const int part = blockIdx.x & 3;
  const int tid  = threadIdx.x;          // 0..511
  const int seg  = tid >> 7;             // gate segment 0..3
  const int jl   = tid & 127;            // local hidden idx
  const int jg   = part * 128 + jl;      // global hidden idx (0..511)
  const int row  = seg * HHD + jg;       // gate row in [0, 2048)

  const float*    G    = dir ? Gb : Gf;
  const _Float16* Whh  = dir ? Whhb16 : Whhf16;
  _Float16*       hb0  = hbuf + (size_t)dir * 2 * HHD;        // buffer parity 0
  _Float16*       hb1  = hb0 + HHD;                           // buffer parity 1
  int*            pcnt = cnt + dir;

  // initial h / c
  float c = 0.0f;
  if (tid < 128) {
    c = c0[dir * HHD + jg];
    hb0[jg] = (_Float16)h0[dir * HHD + jg];
  }
  __threadfence();
  __syncthreads();
  if (tid == 0)
    __hip_atomic_fetch_add(pcnt, 1, __ATOMIC_RELEASE, __HIP_MEMORY_SCOPE_AGENT);

  const v8h* __restrict__ wrow = (const v8h*)(Whh + (size_t)row * HHD);
  const v8h* __restrict__ hv   = (const v8h*)sH;

  for (int s = 0; s < T_LEN; ++s) {
    const int t = dir ? (T_LEN - 1 - s) : s;
    const _Float16* __restrict__ hsrc = (s & 1) ? hb1 : hb0;
    _Float16* __restrict__       hdst = (s & 1) ? hb0 : hb1;

    // wait until all 4 blocks of this direction published h_s
    if (tid == 0) {
      const int target = 4 * (s + 1);
      while (__hip_atomic_load(pcnt, __ATOMIC_ACQUIRE, __HIP_MEMORY_SCOPE_AGENT) < target)
        __builtin_amdgcn_s_sleep(1);
    }
    __syncthreads();

    // stage h_s into LDS (f16, 1KB)
    sH[tid] = hsrc[tid];
    __syncthreads();

    // packed-f16 dot: 512 MACs as 64 x v8h pk-FMA steps, f16x8 accumulator
    v8h acc8 = (v8h){};
#pragma unroll 8
    for (int k = 0; k < HHD / 8; ++k) {
      acc8 = wrow[k] * hv[k] + acc8;      // 4x v_pk_fma_f16
    }
    float dot = 0.0f;
#pragma unroll
    for (int e = 0; e < 8; ++e) dot += (float)acc8[e];

    sG[seg * 128 + jl] = G[(size_t)t * G4 + row] + dot;
    __syncthreads();

    if (tid < 128) {
      const float ig = sG[jl];
      const float fg = sG[128 + jl];
      const float gg = sG[256 + jl];
      const float og = sG[384 + jl];
      const float i_ = 1.0f / (1.0f + __expf(-ig));
      const float f_ = 1.0f / (1.0f + __expf(-fg));
      const float o_ = 1.0f / (1.0f + __expf(-og));
      c = f_ * c + i_ * tanhf(gg);
      const float h = o_ * tanhf(c);
      const _Float16 h16 = (_Float16)h;
      hdst[jg] = h16;
      H[(size_t)t * (2 * HHD) + dir * HHD + jg] = h16;
      __threadfence();
    }
    __syncthreads();
    if (tid == 0)
      __hip_atomic_fetch_add(pcnt, 1, __ATOMIC_RELEASE, __HIP_MEMORY_SCOPE_AGENT);
  }
}

// ---------------------------------------------------------------------------
// WMMA tag projection: feats[T,16] = H[T,1024] @ Wt[16,1024]^T + b_tag
// grid = T/16, block = 32 (one wave, one 16x16 tile per block).
// ---------------------------------------------------------------------------
__global__ void wmma_tag(const _Float16* __restrict__ H,
                         const _Float16* __restrict__ Wt,
                         const float* __restrict__ btag,
                         float* __restrict__ feats) {
  const int m0 = blockIdx.x * 16;
  v8f acc = {};
  for (int k0 = 0; k0 < 2 * HHD; k0 += 32) {
    v16h a = load_frag16(H + (size_t)m0 * (2 * HHD) + k0, 2 * HHD);
    v16h b = load_frag16(Wt + k0, 2 * HHD);
    acc = __builtin_amdgcn_wmma_f32_16x16x32_f16(false, a, false, b, (short)0, acc, false, false);
  }
  const int lane = threadIdx.x & 31;
  const int colL = lane & 15;
  const int rhi  = (lane >> 4) * 8;
  const float bc = btag[colL];
#pragma unroll
  for (int r = 0; r < 8; ++r) {
    const int row = m0 + r + rhi;
    feats[(size_t)row * NTAG + colL] = acc[r] + bc;
  }
}

// ---------------------------------------------------------------------------
// Viterbi decode: single wave. Lane n (<16) owns next-tag n; transitions row
// cached in registers; fv in LDS; serial backtrack by lane 0.
// out[0] = score, out[1..T] = path tags (as floats).
// ---------------------------------------------------------------------------
__global__ void viterbi(const float* __restrict__ feats,
                        const float* __restrict__ trans,
                        int* __restrict__ bptr,
                        float* __restrict__ out) {
  __shared__ float sFv[NTAG];
  __shared__ float sTerm[NTAG];
  const int n = threadIdx.x;
  float tr[NTAG];
  if (n < NTAG) {
#pragma unroll
    for (int p = 0; p < NTAG; ++p) tr[p] = trans[n * NTAG + p];
    sFv[n] = (n == START_TAG) ? 0.0f : NEGV;
  }
  __syncthreads();
  for (int t = 0; t < T_LEN; ++t) {
    float best = -3.0e38f;
    int bi = 0;
    if (n < NTAG) {
#pragma unroll
      for (int p = 0; p < NTAG; ++p) {
        const float v = sFv[p] + tr[p];
        if (v > best) { best = v; bi = p; }   // strict > keeps first max (argmax tie rule)
      }
    }
    __syncthreads();
    if (n < NTAG) {
      sFv[n] = best + feats[t * NTAG + n];
      bptr[t * NTAG + n] = bi;
    }
    __syncthreads();
  }
  if (n < NTAG) {
    float term = sFv[n] + trans[STOP_TAG * NTAG + n];
    if (n == STOP_TAG || n == START_TAG) term = NEGV;
    sTerm[n] = term;
  }
  __syncthreads();
  if (n == 0) {
    int best = 0;
    float bv = sTerm[0];
    for (int p = 1; p < NTAG; ++p)
      if (sTerm[p] > bv) { bv = sTerm[p]; best = p; }
    out[0] = bv;
    int tag = best;
    out[1 + (T_LEN - 1)] = (float)tag;
    for (int t = T_LEN - 2; t >= 0; --t) {
      tag = bptr[(t + 1) * NTAG + tag];
      out[1 + t] = (float)tag;
    }
  }
}

// ---------------------------------------------------------------------------
extern "C" void kernel_launch(void* const* d_in, const int* in_sizes, int n_in,
                              void* d_out, int out_size, void* d_ws, size_t ws_size,
                              hipStream_t stream) {
  const float* X      = (const float*)d_in[0];   // [T, IN]
  const float* h0     = (const float*)d_in[1];   // [2,1,HHD]
  const float* c0     = (const float*)d_in[2];
  const float* w_ih_f = (const float*)d_in[3];   // [G4, IN]
  const float* w_hh_f = (const float*)d_in[4];   // [G4, HHD]
  const float* b_ih_f = (const float*)d_in[5];
  const float* b_hh_f = (const float*)d_in[6];
  const float* w_ih_b = (const float*)d_in[7];
  const float* w_hh_b = (const float*)d_in[8];
  const float* b_ih_b = (const float*)d_in[9];
  const float* b_hh_b = (const float*)d_in[10];
  const float* w_tag  = (const float*)d_in[11];  // [NTAG, 2*HHD]
  const float* b_tag  = (const float*)d_in[12];
  const float* trans  = (const float*)d_in[13];  // [NTAG, NTAG]

  // ---- workspace carve (256B aligned) ----
  size_t off = 0;
  char* base = (char*)d_ws;
  auto carve = [&](size_t bytes) -> void* {
    void* p = base + off;
    off += (bytes + 255) & ~(size_t)255;
    return p;
  };
  _Float16* X16    = (_Float16*)carve((size_t)T_LEN * IN_DIM * 2);
  _Float16* Wf16   = (_Float16*)carve((size_t)G4 * IN_DIM * 2);
  _Float16* Wb16   = (_Float16*)carve((size_t)G4 * IN_DIM * 2);
  _Float16* Whhf16 = (_Float16*)carve((size_t)G4 * HHD * 2);
  _Float16* Whhb16 = (_Float16*)carve((size_t)G4 * HHD * 2);
  _Float16* Wt16   = (_Float16*)carve((size_t)NTAG * 2 * HHD * 2);
  float*    biasf  = (float*)carve((size_t)G4 * 4);
  float*    biasb  = (float*)carve((size_t)G4 * 4);
  float*    Gf     = (float*)carve((size_t)T_LEN * G4 * 4);
  float*    Gb     = (float*)carve((size_t)T_LEN * G4 * 4);
  _Float16* H16    = (_Float16*)carve((size_t)T_LEN * 2 * HHD * 2);
  float*    feats  = (float*)carve((size_t)T_LEN * NTAG * 4);
  int*      bptr   = (int*)carve((size_t)T_LEN * NTAG * 4);
  _Float16* hbuf   = (_Float16*)carve((size_t)2 * 2 * HHD * 2);
  int*      cnt    = (int*)carve(2 * sizeof(int));
  (void)ws_size; (void)in_sizes; (void)n_in; (void)out_size;

  // ---- f32 -> f16 staging + bias sums + sync counter reset ----
  {
    int n;
    n = T_LEN * IN_DIM;
    cvt_f32_f16<<<(n + 255) / 256, 256, 0, stream>>>(X, X16, n);
    n = G4 * IN_DIM;
    cvt_f32_f16<<<(n + 255) / 256, 256, 0, stream>>>(w_ih_f, Wf16, n);
    cvt_f32_f16<<<(n + 255) / 256, 256, 0, stream>>>(w_ih_b, Wb16, n);
    n = G4 * HHD;
    cvt_f32_f16<<<(n + 255) / 256, 256, 0, stream>>>(w_hh_f, Whhf16, n);
    cvt_f32_f16<<<(n + 255) / 256, 256, 0, stream>>>(w_hh_b, Whhb16, n);
    n = NTAG * 2 * HHD;
    cvt_f32_f16<<<(n + 255) / 256, 256, 0, stream>>>(w_tag, Wt16, n);
    add_vec<<<(G4 + 255) / 256, 256, 0, stream>>>(b_ih_f, b_hh_f, biasf, G4);
    add_vec<<<(G4 + 255) / 256, 256, 0, stream>>>(b_ih_b, b_hh_b, biasb, G4);
    zero_ints<<<1, 32, 0, stream>>>(cnt, 2);
  }

  // ---- time-batched input projection GEMM (WMMA, 64x32 per wave) ----
  {
    dim3 grid(G4 / 32, T_LEN / 64, 2);
    wmma_gemm_gates<<<grid, 32, 0, stream>>>(X16, Wf16, Wb16, biasf, biasb, Gf, Gb);
  }

  // ---- serial bidirectional LSTM scan: 4 WGPs per direction, pk-f16 dots ----
  lstm_scan_mb<<<8, 512, 0, stream>>>(Gf, Gb, Whhf16, Whhb16, h0, c0, hbuf, cnt, H16);

  // ---- tag projection (WMMA, N=16 exact tile) ----
  wmma_tag<<<T_LEN / 16, 32, 0, stream>>>(H16, Wt16, b_tag, feats);

  // ---- Viterbi decode + backtrack ----
  viterbi<<<1, 32, 0, stream>>>(feats, trans, bptr, (float*)d_out);
}
